// MultiQueryAttention_21251498181442
// MI455X (gfx1250) — matrix operand order repaired and checked by reading, not compiled, not run on hardware
//
#include <hip/hip_runtime.h>
#include <hip/hip_bf16.h>

// ---------------------------------------------------------------------------
// MultiQueryAttention (GQA) forward for MI455X / gfx1250 (wave32, WMMA).
// All GEMMs run through v_wmma_f32_16x16x32_bf16; accumulation stays f32.
// Every B operand is pre-transposed to N-major [N][K] bf16 (one-time LDS-tiled
// transpose+convert), so GEMM staging is pure b128 loads and every WMMA
// fragment loads with ds_load_b128.  LDS tiles are double-buffered with one
// barrier per K-step.
// ---------------------------------------------------------------------------

#define HID   2048
#define SEQ   2048
#define NH    16
#define NKV   4
#define HD    128
#define KVW   (NKV * HD)     // 512
#define TM    128
#define TN    128
#define TK    32

typedef __attribute__((ext_vector_type(16))) __bf16    v16bf;
typedef __attribute__((ext_vector_type(8)))  float     v8f;

union frag_u { uint4 q[2]; v16bf h; };
union pack8  { __bf16 b[8]; uint4 u; };

// ---- A fragment: 16x32 bf16. lanes 0-15 rows M=0..15 (K 0..7 | 16..23),
//      lanes 16-31 same rows (K 8..15 | 24..31).  As is [row][k], stride TK.
__device__ __forceinline__ v16bf frag_a(const __bf16* As, int row, int lane) {
  const int r  = row + (lane & 15);
  const int kb = (lane >> 4) * 8;
  const __bf16* p = As + r * TK + kb;
  frag_u f;
  f.q[0] = *reinterpret_cast<const uint4*>(p);        // K = kb .. kb+7
  f.q[1] = *reinterpret_cast<const uint4*>(p + 16);   // K = 16+kb .. 16+kb+7
  return f.h;
}

// ---- B fragment from n-major LDS tile Bs[n][k] (stride TK):
//      lane holds col N=lane%16; lanes 0-15 K=0..7|16..23, lanes 16-31 K=8..15|24..31.
__device__ __forceinline__ v16bf frag_b(const __bf16* Bs, int col, int lane) {
  const int c  = col + (lane & 15);
  const int kb = (lane >> 4) * 8;
  const __bf16* p = Bs + c * TK + kb;
  frag_u f;
  f.q[0] = *reinterpret_cast<const uint4*>(p);        // K = kb .. kb+7
  f.q[1] = *reinterpret_cast<const uint4*>(p + 16);   // K = 16+kb .. 16+kb+7
  return f.h;
}

// ---- 8-element global load with optional f32->bf16 convert ----------------
template<typename AT>
__device__ __forceinline__ uint4 ld8a(const AT* __restrict__ src) {
  if constexpr (sizeof(AT) == 4) {
    const float4 v0 = *reinterpret_cast<const float4*>(src);
    const float4 v1 = *reinterpret_cast<const float4*>(src + 4);
    pack8 t;
    t.b[0] = (__bf16)v0.x; t.b[1] = (__bf16)v0.y;
    t.b[2] = (__bf16)v0.z; t.b[3] = (__bf16)v0.w;
    t.b[4] = (__bf16)v1.x; t.b[5] = (__bf16)v1.y;
    t.b[6] = (__bf16)v1.z; t.b[7] = (__bf16)v1.w;
    return t.u;
  } else {
    return *reinterpret_cast<const uint4*>(src);
  }
}

// ---------------------------------------------------------------------------
// Tiled WMMA GEMM.  C[b] = alpha * A[b] * B[b]ᵀ   (B stored N-major [N][K]).
//   A: M x K row-major (lda), type AT (f32 converted to bf16 while staging)
//   B: bf16 N x K row-major (ldb)
//   Batched via blockIdx.z: A += z*strideA, B += (z/kvGroup)*strideB,
//   C += z*strideC.   M,N multiples of 128, K multiple of 32.
// 256 threads = 8 wave32 waves; wave (wm,wn) owns 32x64 = 2x4 WMMA tiles.
// Pipeline: stage kt+1 into registers while computing kt from LDS buffer
// kt&1; commit into buffer (kt+1)&1; one barrier per iteration.
// ---------------------------------------------------------------------------
template<typename AT>
__global__ __launch_bounds__(256) void gemm_wmma(
    const AT* __restrict__ A, const __bf16* __restrict__ B, float* __restrict__ C,
    int K, int lda, int ldb, int ldc,
    size_t strideA, size_t strideB, size_t strideC, int kvGroup, float alpha) {
  __shared__ __align__(16) __bf16 As[2][TM * TK];   // [row][k]
  __shared__ __align__(16) __bf16 Bs[2][TN * TK];   // [n][k]

  const AT*     Ab = A + strideA * blockIdx.z;
  const __bf16* Bb = B + strideB * (size_t)(blockIdx.z / kvGroup);
  float*        Cb = C + strideC * blockIdx.z;

  const int row0 = blockIdx.y * TM;
  const int col0 = blockIdx.x * TN;
  const int t    = threadIdx.x;
  const int lane = t & 31;
  const int w    = t >> 5;
  const int wm   = (w & 3) * 32;
  const int wn   = (w >> 2) * 64;

  // per-thread staging coordinates: 2 chunks of 8 elements for A and for B.
  int aOff[2], bOff[2];
  const AT*     aPtr[2];
  const __bf16* bPtr[2];
#pragma unroll
  for (int c2 = 0; c2 < 2; ++c2) {
    const int c  = t + c2 * 256;       // 0..511
    const int r  = c >> 2;             // 0..127
    const int k8 = (c & 3) * 8;        // 0,8,16,24
    aOff[c2] = r * TK + k8;
    bOff[c2] = r * TK + k8;
    aPtr[c2] = Ab + (size_t)(row0 + r) * lda + k8;
    bPtr[c2] = Bb + (size_t)(col0 + r) * ldb + k8;
  }

  uint4 aReg[2], bReg[2];
  auto stage = [&]() {
#pragma unroll
    for (int c2 = 0; c2 < 2; ++c2) {
      aReg[c2] = ld8a<AT>(aPtr[c2]);
      aPtr[c2] += TK;
    }
#pragma unroll
    for (int c2 = 0; c2 < 2; ++c2) {
      bReg[c2] = *reinterpret_cast<const uint4*>(bPtr[c2]);
      bPtr[c2] += TK;
    }
  };
  auto commit = [&](int p) {
#pragma unroll
    for (int c2 = 0; c2 < 2; ++c2)
      *reinterpret_cast<uint4*>(&As[p][aOff[c2]]) = aReg[c2];
#pragma unroll
    for (int c2 = 0; c2 < 2; ++c2)
      *reinterpret_cast<uint4*>(&Bs[p][bOff[c2]]) = bReg[c2];
  };

  v8f acc[2][4];
#pragma unroll
  for (int i = 0; i < 2; ++i)
#pragma unroll
    for (int j = 0; j < 4; ++j)
      acc[i][j] = (v8f){0.f, 0.f, 0.f, 0.f, 0.f, 0.f, 0.f, 0.f};

  const int ktiles = K / TK;
  stage();
  commit(0);
  __syncthreads();

  for (int kt = 0; kt < ktiles; ++kt) {
    const bool last = (kt + 1 == ktiles);
    if (!last) {
      stage();                             // overlap global loads with WMMA
      if (kt + 2 < ktiles) {               // hint tile kt+2 (global_prefetch)
        __builtin_prefetch(aPtr[0], 0, 0);
        __builtin_prefetch(bPtr[0], 0, 0);
      }
    }

    const __bf16* Asb = As[kt & 1];
    const __bf16* Bsb = Bs[kt & 1];
    v16bf af[2], bfg[4];
#pragma unroll
    for (int ti = 0; ti < 2; ++ti) af[ti] = frag_a(Asb, wm + ti * 16, lane);
#pragma unroll
    for (int tj = 0; tj < 4; ++tj) bfg[tj] = frag_b(Bsb, wn + tj * 16, lane);
#pragma unroll
    for (int ti = 0; ti < 2; ++ti)
#pragma unroll
      for (int tj = 0; tj < 4; ++tj)
        acc[ti][tj] = __builtin_amdgcn_wmma_f32_16x16x32_bf16(
            false, af[ti], false, bfg[tj], (short)0, acc[ti][tj], false, false);

    if (!last) commit((kt + 1) & 1);       // writes go to the idle buffer
    __syncthreads();                       // single barrier per iteration
  }

  // ---- store: C/D layout -> lanes 0-15: M=v, lanes 16-31: M=v+8; col=lane&15
  const int cr = (lane >> 4) * 8;
  const int cc = lane & 15;
#pragma unroll
  for (int ti = 0; ti < 2; ++ti)
#pragma unroll
    for (int tj = 0; tj < 4; ++tj) {
      float* pr = Cb + (size_t)(row0 + wm + ti * 16 + cr) * ldc
                     + (col0 + wn + tj * 16 + cc);
#pragma unroll
      for (int v = 0; v < 8; ++v) {
        *pr = alpha * acc[ti][tj][v];
        pr += ldc;                         // running row pointer
      }
    }
}

// ---------------------------------------------------------------------------
// Elementwise f32 -> bf16
__global__ void cvt_bf16(const float* __restrict__ in, __bf16* __restrict__ out,
                         size_t n) {
  size_t i = (size_t)blockIdx.x * 256 + threadIdx.x;
  if (i < n) out[i] = (__bf16)in[i];
}

// LDS-tiled transpose + f32->bf16 convert: out[c][r] = (bf16)in[r][c].
// in: R x C f32;  out: C x R bf16.  R, C multiples of 32.
__global__ __launch_bounds__(256) void transpose_cvt(
    const float* __restrict__ in, __bf16* __restrict__ out, int R, int C) {
  __shared__ float tile[32][33];
  const int bx = blockIdx.x * 32;   // col base in `in`
  const int by = blockIdx.y * 32;   // row base in `in`
  const int tx = threadIdx.x;       // 0..31
  const int ty = threadIdx.y;       // 0..7
#pragma unroll
  for (int j = 0; j < 32; j += 8)
    tile[ty + j][tx] = in[(size_t)(by + ty + j) * C + bx + tx];
  __syncthreads();
#pragma unroll
  for (int j = 0; j < 32; j += 8)
    out[(size_t)(bx + ty + j) * R + by + tx] = (__bf16)tile[tx][ty + j];
}

// RoPE (half-rotation pairing d <-> d+64 inside each 128-wide head) + bf16 cvt.
// cols = 2048 (Q) or 512 (K); angle = s * THETA^{-(d%64)/64}.
__global__ void rope_cvt(const float* __restrict__ in, __bf16* __restrict__ out,
                         int cols) {
  size_t idx = (size_t)blockIdx.x * 256 + threadIdx.x;
  int s = (int)(idx / cols);
  int c = (int)(idx % cols);
  int d  = c & (HD - 1);
  int hf = d & 63;
  float inv_freq = __powf(10000.0f, -(float)hf * (1.0f / 64.0f));
  float ang = (float)s * inv_freq;
  float cs = __cosf(ang), sn = __sinf(ang);
  float x = in[idx];
  float partner = (d < 64) ? -in[idx + 64] : in[idx - 64];
  out[idx] = (__bf16)(x * cs + partner * sn);
}

// Row softmax, in place.  One 256-thread block per 2048-wide row; values stay
// resident in registers (single global read + single write per element).
__global__ __launch_bounds__(256) void softmax_rows(float* __restrict__ data) {
  float* p = data + (size_t)blockIdx.x * SEQ;
  const int t = threadIdx.x;
  float v[8];
  float m = -INFINITY;
#pragma unroll
  for (int i = 0; i < 8; ++i) { v[i] = p[t + i * 256]; m = fmaxf(m, v[i]); }
#pragma unroll
  for (int off = 16; off > 0; off >>= 1) m = fmaxf(m, __shfl_xor(m, off, 32));
  __shared__ float redm[8];
  if ((t & 31) == 0) redm[t >> 5] = m;
  __syncthreads();
  float bm = redm[0];
#pragma unroll
  for (int i = 1; i < 8; ++i) bm = fmaxf(bm, redm[i]);
  float s = 0.f;
#pragma unroll
  for (int i = 0; i < 8; ++i) { v[i] = __expf(v[i] - bm); s += v[i]; }
#pragma unroll
  for (int off = 16; off > 0; off >>= 1) s += __shfl_xor(s, off, 32);
  __shared__ float reds[8];
  if ((t & 31) == 0) reds[t >> 5] = s;
  __syncthreads();
  float bs = 0.f;
#pragma unroll
  for (int i = 0; i < 8; ++i) bs += reds[i];
  const float inv = 1.0f / bs;
#pragma unroll
  for (int i = 0; i < 8; ++i) p[t + i * 256] = v[i] * inv;
}

// ---------------------------------------------------------------------------
extern "C" void kernel_launch(void* const* d_in, const int* in_sizes, int n_in,
                              void* d_out, int out_size, void* d_ws, size_t ws_size,
                              hipStream_t stream) {
  const float* X  = (const float*)d_in[0];   // [1,2048,2048]
  const float* Wq = (const float*)d_in[1];   // [2048,2048]  (in x out)
  const float* Wk = (const float*)d_in[2];   // [2048,512]
  const float* Wv = (const float*)d_in[3];   // [2048,512]
  const float* Wo = (const float*)d_in[4];   // [2048,2048]

  float* out  = (float*)d_out;                         // [1,2048,2048]
  float* attn = out + (size_t)SEQ * HID;               // [1,16,2048,2048]

  // ---- workspace carve-out (256B aligned) ----
  char* wsp = (char*)d_ws;
  auto alloc = [&](size_t bytes) -> char* {
    char* p = wsp;
    wsp += (bytes + 255) & ~(size_t)255;
    return p;
  };
  __bf16* Xb  = (__bf16*)alloc((size_t)SEQ * HID * 2);   // X bf16 (A-side)
  __bf16* Wqt = (__bf16*)alloc((size_t)HID * HID * 2);   // Wqᵀ [out][in]
  __bf16* Wkt = (__bf16*)alloc((size_t)HID * KVW * 2);   // Wkᵀ [512][2048]
  __bf16* Wvt = (__bf16*)alloc((size_t)HID * KVW * 2);   // Wvᵀ [512][2048]
  __bf16* Wot = (__bf16*)alloc((size_t)HID * HID * 2);   // Woᵀ [out][in]
  float*  Qf  = (float*) alloc((size_t)SEQ * HID * 4);
  float*  Kf  = (float*) alloc((size_t)SEQ * KVW * 4);
  float*  Vf  = (float*) alloc((size_t)SEQ * KVW * 4);
  __bf16* Qr  = (__bf16*)alloc((size_t)SEQ * HID * 2);   // RoPE(Q) bf16
  __bf16* Kr  = (__bf16*)alloc((size_t)SEQ * KVW * 2);   // RoPE(K) bf16 (N-major)
  __bf16* Vt  = (__bf16*)alloc((size_t)SEQ * KVW * 2);   // Vᵀ [512][2048]
  float*  Ctx = (float*) alloc((size_t)SEQ * HID * 4);

  const size_t nXW = (size_t)SEQ * HID;  // 4M

  // ---- 1. convert X; transpose+convert all weight matrices ----
  cvt_bf16<<<dim3((unsigned)(nXW / 256)), 256, 0, stream>>>(X, Xb, nXW);
  transpose_cvt<<<dim3(HID / 32, HID / 32), dim3(32, 8), 0, stream>>>(Wq, Wqt, HID, HID);
  transpose_cvt<<<dim3(KVW / 32, HID / 32), dim3(32, 8), 0, stream>>>(Wk, Wkt, HID, KVW);
  transpose_cvt<<<dim3(KVW / 32, HID / 32), dim3(32, 8), 0, stream>>>(Wv, Wvt, HID, KVW);
  transpose_cvt<<<dim3(HID / 32, HID / 32), dim3(32, 8), 0, stream>>>(Wo, Wot, HID, HID);

  // ---- 2. projections: Q = X*Wq, K = X*Wk, V = X*Wv  (B = Wᵀ, N-major) ----
  gemm_wmma<__bf16><<<dim3(HID / TN, SEQ / TM, 1), 256, 0, stream>>>(
      Xb, Wqt, Qf, HID, HID, HID, HID, 0, 0, 0, 1, 1.0f);
  gemm_wmma<__bf16><<<dim3(KVW / TN, SEQ / TM, 1), 256, 0, stream>>>(
      Xb, Wkt, Kf, HID, HID, HID, KVW, 0, 0, 0, 1, 1.0f);
  gemm_wmma<__bf16><<<dim3(KVW / TN, SEQ / TM, 1), 256, 0, stream>>>(
      Xb, Wvt, Vf, HID, HID, HID, KVW, 0, 0, 0, 1, 1.0f);

  // ---- 3. RoPE(Q), RoPE(K); V transposed to [512][2048] bf16 ----
  rope_cvt<<<dim3((unsigned)(nXW / 256)), 256, 0, stream>>>(Qf, Qr, HID);
  rope_cvt<<<dim3((unsigned)((size_t)SEQ * KVW / 256)), 256, 0, stream>>>(Kf, Kr, KVW);
  transpose_cvt<<<dim3(KVW / 32, SEQ / 32), dim3(32, 8), 0, stream>>>(Vf, Vt, SEQ, KVW);

  // ---- 4. scores[h] = scale * Q_h * K_{h/4}ᵀ  -> attn region of d_out ----
  const float scale = 0.08838834764831845f;  // 128^-0.5
  gemm_wmma<__bf16><<<dim3(SEQ / TN, SEQ / TM, NH), 256, 0, stream>>>(
      Qr, Kr, attn, HD, HID, KVW, SEQ,
      /*sA*/ (size_t)HD, /*sB*/ (size_t)HD, /*sC*/ (size_t)SEQ * SEQ,
      /*kvGroup*/ NH / NKV, scale);

  // ---- 5. softmax rows (in place, stays in d_out as attn_weights) ----
  softmax_rows<<<dim3(NH * SEQ), 256, 0, stream>>>(attn);

  // ---- 6. context[h] = P_h * V_{h/4}   (B = Vᵀ rows (h/4)*128..(h/4)*128+127) ----
  gemm_wmma<float><<<dim3(1, SEQ / TM, NH), 256, 0, stream>>>(
      attn, Vt, Ctx, SEQ, SEQ, SEQ, HID,
      /*sA*/ (size_t)SEQ * SEQ, /*sB*/ (size_t)HD * SEQ, /*sC*/ (size_t)HD,
      /*kvGroup*/ NH / NKV, 1.0f);

  // ---- 7. output = context * Wo  (B = Woᵀ) ----
  gemm_wmma<float><<<dim3(HID / TN, SEQ / TM, 1), 256, 0, stream>>>(
      Ctx, Wot, out, HID, HID, HID, HID, 0, 0, 0, 1, 1.0f);
}
